// BaseConvolution_26353919328303
// MI455X (gfx1250) — compile-verified, matrix operand order checked by reading, not run on hardware
//
#include <hip/hip_runtime.h>
#include <hip/hip_bf16.h>
#include <stdint.h>

// ---- problem constants (match reference) ----
#define N_PTS   16384
#define F_INF   64
#define M_CTR   4096
#define KNBR    64
#define F_HID   64
#define F_OUT   128
#define RADIUS2 0.04f      // 0.2^2
#define CAP     1024       // max in-radius candidates kept per center

// ---- LDS strides (bf16 elements); odd-dword byte strides -> conflict-free ----
#define AST   98           // activation buffers, width up to 96
#define WST1  98           // W1t rows (width 96)
#define WST   66           // W2t/W3t rows (width 64)

typedef __bf16 bf16;
typedef __attribute__((ext_vector_type(16))) __bf16 v16bf;
typedef __attribute__((ext_vector_type(8)))  float  v8f;

// Load one 16x32 bf16 A/B fragment per the CDNA5 ISA layout:
// lane L holds row (L&15); khalf=L>>4; VGPR j<4: K=khalf*8+2j(+1); j>=4: K=16+khalf*8+...
__device__ __forceinline__ v16bf load_frag16(const bf16* __restrict__ base,
                                             int row, int stride, int k0, int lane) {
  const bf16* p = base + row * stride + k0 + ((lane >> 4) << 3);
  v16bf f;
#pragma unroll
  for (int j = 0; j < 4; ++j) { f[2*j]   = p[2*j];      f[2*j+1]   = p[2*j+1]; }
#pragma unroll
  for (int j = 0; j < 4; ++j) { f[8+2*j] = p[16+2*j];   f[8+2*j+1] = p[16+2*j+1]; }
  return f;
}

// =============== 1) weight prep: fp32 -> transposed, padded bf16 ===============
__global__ __launch_bounds__(256) void k_prep(const float* __restrict__ W1,
                                              const float* __restrict__ W2,
                                              const float* __restrict__ W3,
                                              bf16* __restrict__ w1t,
                                              bf16* __restrict__ w2t,
                                              bf16* __restrict__ w3t) {
  const int tid = threadIdx.x;
  for (int e = tid; e < 64 * WST1; e += 256) {      // W1t[n][k], n<64, k<98 (67 real)
    int n = e / WST1, k = e % WST1;
    float v = (k < F_INF + 3) ? W1[k * F_HID + n] : 0.f;
    w1t[e] = (bf16)v;
  }
  for (int e = tid; e < 64 * WST; e += 256) {       // W2t[n][k]
    int n = e / WST, k = e % WST;
    float v = (k < F_HID) ? W2[k * F_HID + n] : 0.f;
    w2t[e] = (bf16)v;
  }
  for (int e = tid; e < 128 * WST; e += 256) {      // W3t[n][k]
    int n = e / WST, k = e % WST;
    float v = (k < F_HID) ? W3[k * F_OUT + n] : 0.f;
    w3t[e] = (bf16)v;
  }
}

// =============== 2) farthest point sampling (single workgroup, pos in LDS) ===============
__global__ __launch_bounds__(1024) void k_fps(const float* __restrict__ pos,
                                              int* __restrict__ idx) {
  extern __shared__ char smem[];
  float* px  = (float*)smem;
  float* py  = px + N_PTS;
  float* pz  = py + N_PTS;
  float* sRv = pz + N_PTS;              // 32 wave partial values
  int*   sRi = (int*)(sRv + 32);        // 32 wave partial indices
  int*   sCur = sRi + 32;

  const int tid = threadIdx.x;
  for (int p = tid; p < N_PTS; p += 1024) {
    px[p] = pos[p * 3 + 0]; py[p] = pos[p * 3 + 1]; pz[p] = pos[p * 3 + 2];
  }
  if (tid == 0) *sCur = 0;
  float md[16];
#pragma unroll
  for (int j = 0; j < 16; ++j) md[j] = __builtin_inff();
  __syncthreads();

  for (int it = 0; it < M_CTR; ++it) {
    const int cur = *sCur;
    if (tid == 0) idx[it] = cur;
    const float cx = px[cur], cy = py[cur], cz = pz[cur];
    float bv = -1.f; int bi = 0;
#pragma unroll
    for (int j = 0; j < 16; ++j) {
      const int p = tid + j * 1024;
      const float dx = px[p] - cx, dy = py[p] - cy, dz = pz[p] - cz;
      const float d2 = dx * dx + dy * dy + dz * dz;
      md[j] = md[j] < d2 ? md[j] : d2;
      if (md[j] > bv) { bv = md[j]; bi = p; }
    }
    for (int o = 16; o > 0; o >>= 1) {               // wave32 argmax
      float ov = __shfl_down(bv, o, 32);
      int   oi = __shfl_down(bi, o, 32);
      if (ov > bv || (ov == bv && oi < bi)) { bv = ov; bi = oi; }
    }
    if ((tid & 31) == 0) { sRv[tid >> 5] = bv; sRi[tid >> 5] = bi; }
    __syncthreads();
    if (tid < 32) {
      bv = sRv[tid]; bi = sRi[tid];
      for (int o = 16; o > 0; o >>= 1) {
        float ov = __shfl_down(bv, o, 32);
        int   oi = __shfl_down(bi, o, 32);
        if (ov > bv || (ov == bv && oi < bi)) { bv = ov; bi = oi; }
      }
      if (tid == 0) *sCur = bi;
    }
    __syncthreads();
  }
}

// =============== 3) gather sampled positions / batch ===============
__global__ __launch_bounds__(256) void k_gather(const int* __restrict__ idx,
                                                const float* __restrict__ pos,
                                                const int* __restrict__ batch,
                                                float* __restrict__ pos_s,
                                                int* __restrict__ batch_s) {
  const int i = blockIdx.x * 256 + threadIdx.x;
  if (i < M_CTR) {
    const int id = idx[i];
    pos_s[i * 3 + 0] = pos[id * 3 + 0];
    pos_s[i * 3 + 1] = pos[id * 3 + 1];
    pos_s[i * 3 + 2] = pos[id * 3 + 2];
    batch_s[i] = batch[id];
  }
}

// =============== 4) k-nearest within radius (per-center workgroup) ===============
// nbrp[m*K + s]: neighbor point index; bit31 set => invalid slot (index bits then 0).
__global__ __launch_bounds__(256) void k_nbr(const float* __restrict__ pos,
                                             const int* __restrict__ batch,
                                             const float* __restrict__ pos_s,
                                             const int* __restrict__ batch_s,
                                             int* __restrict__ nbrp) {
  __shared__ float cd2[CAP];
  __shared__ int   cidx[CAP];
  __shared__ int   cnt;
  __shared__ float sRv[8];
  __shared__ int   sRi[8];
  const int m = blockIdx.x, tid = threadIdx.x;
  if (tid == 0) cnt = 0;
  __syncthreads();

  const float cx = pos_s[m * 3 + 0], cy = pos_s[m * 3 + 1], cz = pos_s[m * 3 + 2];
  const int bm = batch_s[m];
  for (int p = tid; p < N_PTS; p += 256) {
    const float dx = pos[p * 3 + 0] - cx, dy = pos[p * 3 + 1] - cy, dz = pos[p * 3 + 2] - cz;
    const float d2 = dx * dx + dy * dy + dz * dz;
    if (d2 <= RADIUS2 && batch[p] == bm) {
      const int s = atomicAdd(&cnt, 1);
      if (s < CAP) { cd2[s] = d2; cidx[s] = p; }
    }
  }
  __syncthreads();
  const int C = cnt < CAP ? cnt : CAP;
  int* outp = nbrp + (size_t)m * KNBR;
  if (C <= KNBR) {
    if (tid < KNBR) outp[tid] = (tid < C) ? cidx[tid] : (int)0x80000000;
  } else {
    for (int s = 0; s < KNBR; ++s) {                 // iterative argmin selection
      float bv = __builtin_inff(); int bs = 0;
      for (int j = tid; j < C; j += 256)
        if (cd2[j] < bv) { bv = cd2[j]; bs = j; }
      for (int o = 16; o > 0; o >>= 1) {
        float ov = __shfl_down(bv, o, 32);
        int   oi = __shfl_down(bs, o, 32);
        if (ov < bv) { bv = ov; bs = oi; }
      }
      if ((tid & 31) == 0) { sRv[tid >> 5] = bv; sRi[tid >> 5] = bs; }
      __syncthreads();
      if (tid == 0) {
        float v = sRv[0]; int sl = sRi[0];
        for (int w = 1; w < 8; ++w) if (sRv[w] < v) { v = sRv[w]; sl = sRi[w]; }
        outp[s] = cidx[sl];
        cd2[sl] = __builtin_inff();
      }
      __syncthreads();
    }
  }
}

// =============== 5) per-edge MLP (bf16 WMMA) + masked max aggregation ===============
__global__ __launch_bounds__(256) void k_mlp(const float* __restrict__ x,
                                             const float* __restrict__ pos,
                                             const float* __restrict__ pos_s,
                                             const int* __restrict__ nbrp,
                                             const bf16* __restrict__ gW1,
                                             const bf16* __restrict__ gW2,
                                             const bf16* __restrict__ gW3,
                                             const float* __restrict__ gb1,
                                             const float* __restrict__ gb2,
                                             const float* __restrict__ gb3,
                                             float* __restrict__ out) {
  extern __shared__ char smem[];
  bf16*  sA    = (bf16*)smem;            // 64*98 activations (H0 / H2)
  bf16*  sB    = sA + 64 * AST;          // 64*98 activations (H1)
  bf16*  sW1   = sB + 64 * AST;          // 64*98
  bf16*  sW2   = sW1 + 64 * WST1;        // 64*66
  bf16*  sW3   = sW2 + 64 * WST;         // 128*66
  float* sBias = (float*)(sW3 + 128 * WST); // b1[64] b2[64] b3[128]
  int*   sMax  = (int*)(sBias + 256);    // 128
  int*   sNbr  = sMax + 128;             // 64 packed neighbor slots

  const int m = blockIdx.x, tid = threadIdx.x;
  const int lane = tid & 31, wid = tid >> 5;

  // ---- stage weights / biases / neighbor list into LDS ----
  { const uint32_t* g = (const uint32_t*)gW1; uint32_t* l = (uint32_t*)sW1;
    for (int i = tid; i < 64 * WST1 / 2; i += 256) l[i] = g[i]; }
  { const uint32_t* g = (const uint32_t*)gW2; uint32_t* l = (uint32_t*)sW2;
    for (int i = tid; i < 64 * WST / 2; i += 256) l[i] = g[i]; }
  { const uint32_t* g = (const uint32_t*)gW3; uint32_t* l = (uint32_t*)sW3;
    for (int i = tid; i < 128 * WST / 2; i += 256) l[i] = g[i]; }
  if (tid < 64)  sBias[tid]       = gb1[tid];
  if (tid < 64)  sBias[64 + tid]  = gb2[tid];
  if (tid < 128) sBias[128 + tid] = gb3[tid];
  if (tid < 128) sMax[tid] = 0;                      // post-ReLU max >= 0, exact
  if (tid < 64)  sNbr[tid] = nbrp[(size_t)m * KNBR + tid];
  __syncthreads();

  // ---- gather H0 = [x[nbr] | rel | 0-pad] into sA (bf16) ----
  {
    const float cx = pos_s[m * 3 + 0], cy = pos_s[m * 3 + 1], cz = pos_s[m * 3 + 2];
    const int r = tid >> 2, part = tid & 3;          // 4 threads per row, 24 cols each
    const int nb = sNbr[r] & 0x7FFFFFFF;
    const float* xr = x + (size_t)nb * F_INF;
#pragma unroll
    for (int q = 0; q < 24; ++q) {
      const int c = part * 24 + q;
      float v;
      if (c < F_INF)      v = xr[c];
      else if (c == 64)   v = pos[nb * 3 + 0] - cx;
      else if (c == 65)   v = pos[nb * 3 + 1] - cy;
      else if (c == 66)   v = pos[nb * 3 + 2] - cz;
      else                v = 0.f;
      sA[r * AST + c] = (bf16)v;
    }
  }
  __syncthreads();

  // ---- layer 1: H1 = relu(H0[64x96] @ W1 + b1) -> sB ; 16 tiles, 2/wave ----
#pragma unroll
  for (int t = 0; t < 2; ++t) {
    const int tile = wid * 2 + t, tm = tile >> 2, tn = tile & 3;
    const int arow = tm * 16 + (lane & 15), brow = tn * 16 + (lane & 15);
    v8f acc = {0.f, 0.f, 0.f, 0.f, 0.f, 0.f, 0.f, 0.f};
#pragma unroll
    for (int k0 = 0; k0 < 96; k0 += 32) {
      v16bf a = load_frag16(sA,  arow, AST,  k0, lane);
      v16bf b = load_frag16(sW1, brow, WST1, k0, lane);
      acc = __builtin_amdgcn_wmma_f32_16x16x32_bf16(false, a, false, b, (short)0, acc, false, false);
    }
    const int col = tn * 16 + (lane & 15);
    const int rb  = tm * 16 + ((lane >> 4) << 3);
    const float bias = sBias[col];
#pragma unroll
    for (int j = 0; j < 8; ++j) {
      float h = acc[j] + bias; h = h > 0.f ? h : 0.f;
      sB[(rb + j) * AST + col] = (bf16)h;
    }
  }
  __syncthreads();

  // ---- layer 2: H2 = relu(H1[64x64] @ W2 + b2) -> sA ----
#pragma unroll
  for (int t = 0; t < 2; ++t) {
    const int tile = wid * 2 + t, tm = tile >> 2, tn = tile & 3;
    const int arow = tm * 16 + (lane & 15), brow = tn * 16 + (lane & 15);
    v8f acc = {0.f, 0.f, 0.f, 0.f, 0.f, 0.f, 0.f, 0.f};
#pragma unroll
    for (int k0 = 0; k0 < 64; k0 += 32) {
      v16bf a = load_frag16(sB,  arow, AST, k0, lane);
      v16bf b = load_frag16(sW2, brow, WST, k0, lane);
      acc = __builtin_amdgcn_wmma_f32_16x16x32_bf16(false, a, false, b, (short)0, acc, false, false);
    }
    const int col = tn * 16 + (lane & 15);
    const int rb  = tm * 16 + ((lane >> 4) << 3);
    const float bias = sBias[64 + col];
#pragma unroll
    for (int j = 0; j < 8; ++j) {
      float h = acc[j] + bias; h = h > 0.f ? h : 0.f;
      sA[(rb + j) * AST + col] = (bf16)h;
    }
  }
  __syncthreads();

  // ---- layer 3: H3 = relu(H2[64x64] @ W3 + b3), masked max over rows ----
#pragma unroll
  for (int t = 0; t < 4; ++t) {
    const int tile = wid * 4 + t, tm = tile >> 3, tn = tile & 7;
    const int arow = tm * 16 + (lane & 15), brow = tn * 16 + (lane & 15);
    v8f acc = {0.f, 0.f, 0.f, 0.f, 0.f, 0.f, 0.f, 0.f};
#pragma unroll
    for (int k0 = 0; k0 < 64; k0 += 32) {
      v16bf a = load_frag16(sA,  arow, AST, k0, lane);
      v16bf b = load_frag16(sW3, brow, WST, k0, lane);
      acc = __builtin_amdgcn_wmma_f32_16x16x32_bf16(false, a, false, b, (short)0, acc, false, false);
    }
    const int col = tn * 16 + (lane & 15);
    const int rb  = tm * 16 + ((lane >> 4) << 3);
    const float bias = sBias[128 + col];
#pragma unroll
    for (int j = 0; j < 8; ++j) {
      const int row = rb + j;
      if (!(sNbr[row] & 0x80000000)) {               // valid neighbor only
        float h = acc[j] + bias; h = h > 0.f ? h : 0.f;
        atomicMax(&sMax[col], __float_as_int(h));    // exact for non-negative floats
      }
    }
  }
  __syncthreads();
  if (tid < 128) out[(size_t)m * F_OUT + tid] = __int_as_float(sMax[tid]);
}

// =============== launch ===============
extern "C" void kernel_launch(void* const* d_in, const int* in_sizes, int n_in,
                              void* d_out, int out_size, void* d_ws, size_t ws_size,
                              hipStream_t stream) {
  const float* x     = (const float*)d_in[0];
  const float* pos   = (const float*)d_in[1];
  const int*   batch = (const int*)  d_in[2];
  const float* W1    = (const float*)d_in[3];
  const float* b1    = (const float*)d_in[4];
  const float* W2    = (const float*)d_in[5];
  const float* b2    = (const float*)d_in[6];
  const float* W3    = (const float*)d_in[7];
  const float* b3    = (const float*)d_in[8];

  char* ws   = (char*)d_ws;
  int*  idx  = (int*)ws;                                     // 4096 ints
  int*  nbrp = (int*)(ws + 16384);                           // 4096*64 ints
  bf16* w1t  = (bf16*)(ws + 16384 + (size_t)M_CTR * KNBR * 4);
  bf16* w2t  = w1t + 64 * WST1;
  bf16* w3t  = w2t + 64 * WST;

  float* out   = (float*)d_out;                              // [4096,128]
  float* pos_s = out + (size_t)M_CTR * F_OUT;                // [4096,3]
  int*   bat_s = (int*)(pos_s + (size_t)M_CTR * 3);          // [4096]

  hipLaunchKernelGGL(k_prep, dim3(1), dim3(256), 0, stream, W1, W2, W3, w1t, w2t, w3t);

  const size_t fps_lds = (size_t)3 * N_PTS * 4 + 32 * 8 + 16;
  hipLaunchKernelGGL(k_fps, dim3(1), dim3(1024), fps_lds, stream, pos, idx);

  hipLaunchKernelGGL(k_gather, dim3((M_CTR + 255) / 256), dim3(256), 0, stream,
                     idx, pos, batch, pos_s, bat_s);

  hipLaunchKernelGGL(k_nbr, dim3(M_CTR), dim3(256), 0, stream,
                     pos, batch, pos_s, bat_s, nbrp);

  const size_t mlp_lds = (size_t)(64 * AST * 2 + 64 * WST1 + 64 * WST + 128 * WST) * 2
                       + 256 * 4 + 128 * 4 + 64 * 4;
  hipLaunchKernelGGL(k_mlp, dim3(M_CTR), dim3(256), mlp_lds, stream,
                     x, pos, pos_s, nbrp, w1t, w2t, w3t, b1, b2, b3, out);
}
